// ContrastiveLoss_11940009082980
// MI455X (gfx1250) — compile-verified, matrix operand order checked by reading, not run on hardware
//
#include <hip/hip_runtime.h>
#include <stdint.h>

typedef __attribute__((ext_vector_type(16))) _Float16 v16h;
typedef __attribute__((ext_vector_type(8)))  float    v8f;

#define B_    8
#define C_    512
#define HW_   65536
#define QW_   (HW_ / 4)       // 16384 pixel-quads per image
#define K_    64
#define EPS_  1e-8f
#define MARG_ 0.2f

// ---------------------------------------------------------------------------
// K1: per-pixel inverse L2 norm (4 pixels/thread, float4 streams) + counts
// ---------------------------------------------------------------------------
__global__ void k_norm_count(const float* __restrict__ feat,
                             const int* __restrict__ inst,
                             float* __restrict__ invnorm,
                             int* __restrict__ icounts) {
  int t = blockIdx.x * blockDim.x + threadIdx.x;   // quad id [0, B*QW_)
  int b = t >> 14;
  int q = t & (QW_ - 1);
  const float4* fp = (const float4*)feat + (size_t)b * C_ * QW_ + q;
  float s0 = 0.f, s1 = 0.f, s2 = 0.f, s3 = 0.f;
#pragma unroll 4
  for (int c = 0; c < C_; ++c) {
    __builtin_prefetch(fp + (size_t)(c + 16) * QW_, 0, 1);
    float4 x = fp[(size_t)c * QW_];
    s0 = fmaf(x.x, x.x, s0);
    s1 = fmaf(x.y, x.y, s1);
    s2 = fmaf(x.z, x.z, s2);
    s3 = fmaf(x.w, x.w, s3);
  }
  float4 r;
  r.x = 1.0f / fmaxf(sqrtf(s0), EPS_);
  r.y = 1.0f / fmaxf(sqrtf(s1), EPS_);
  r.z = 1.0f / fmaxf(sqrtf(s2), EPS_);
  r.w = 1.0f / fmaxf(sqrtf(s3), EPS_);
  ((float4*)invnorm)[t] = r;
  int4 m4 = ((const int4*)inst)[t];
  atomicAdd(&icounts[b * K_ + m4.x], 1);
  atomicAdd(&icounts[b * K_ + m4.y], 1);
  atomicAdd(&icounts[b * K_ + m4.z], 1);
  atomicAdd(&icounts[b * K_ + m4.w], 1);
}

// ---------------------------------------------------------------------------
// K2: exclusive scan of segment counts -> per-(image,segment) list offsets
// ---------------------------------------------------------------------------
__global__ void k_offsets(const int* __restrict__ icounts, int* __restrict__ offs) {
  int b = blockIdx.x;
  if (threadIdx.x == 0) {
    int acc = 0;
    for (int s = 0; s < K_; ++s) { offs[b * K_ + s] = acc; acc += icounts[b * K_ + s]; }
  }
}

// ---------------------------------------------------------------------------
// K3: compact pixel indices per segment; record each pixel's rank
// ---------------------------------------------------------------------------
__global__ void k_build_lists(const int* __restrict__ inst,
                              const int* __restrict__ offs,
                              int* __restrict__ cursor,
                              int* __restrict__ rank,
                              int* __restrict__ list) {
  int pix = blockIdx.x * blockDim.x + threadIdx.x;
  int b = pix >> 16;
  int n = pix & (HW_ - 1);
  int m = inst[pix];
  int r = atomicAdd(&cursor[b * K_ + m], 1);
  rank[pix] = r;
  list[(size_t)b * HW_ + offs[b * K_ + m] + r] = n;
}

// ---------------------------------------------------------------------------
// K4: segment sums of normalized features (float4 streams into LDS ds_add),
//     flushed with global atomics.
// ---------------------------------------------------------------------------
#define CHUNK_ 128
#define PIXT_  1024
__global__ void k_segsum(const float* __restrict__ feat,
                         const int* __restrict__ inst,
                         const float* __restrict__ invnorm,
                         float* __restrict__ segsum) {
  __shared__ float acc[K_ * CHUNK_];                 // 32 KB
  int tile  = blockIdx.x;
  int pt    = tile & 63;          // 64 pixel tiles per image
  int chunk = (tile >> 6) & 3;    // 4 channel chunks
  int b     = tile >> 8;
  for (int i = threadIdx.x; i < K_ * CHUNK_; i += blockDim.x) acc[i] = 0.f;
  __syncthreads();

  int q0 = pt * (PIXT_ / 4) + threadIdx.x;           // quad index within image
  int4   m4 = ((const int4*)inst)[b * QW_ + q0];
  float4 iv = ((const float4*)invnorm)[b * QW_ + q0];
  const float4* fb = (const float4*)feat + (size_t)b * C_ * QW_ + q0;

  for (int c = 0; c < CHUNK_; ++c) {
    size_t coff = (size_t)(chunk * CHUNK_ + c) * QW_;
    __builtin_prefetch(fb + coff + 8 * (size_t)QW_, 0, 1);
    float4 x = fb[coff];
    atomicAdd(&acc[m4.x * CHUNK_ + c], x.x * iv.x);  // ds_add_f32
    atomicAdd(&acc[m4.y * CHUNK_ + c], x.y * iv.y);
    atomicAdd(&acc[m4.z * CHUNK_ + c], x.z * iv.z);
    atomicAdd(&acc[m4.w * CHUNK_ + c], x.w * iv.w);
  }
  __syncthreads();
  float* gout = segsum + (size_t)b * K_ * C_ + chunk * CHUNK_;
  for (int i = threadIdx.x; i < K_ * CHUNK_; i += blockDim.x) {
    int s = i / CHUNK_, c = i % CHUNK_;
    float v = acc[i];
    if (v != 0.f) atomicAdd(&gout[(size_t)s * C_ + c], v);
  }
}

// ---------------------------------------------------------------------------
// K5: intra term — cyclic within-segment permutation partner, cosine sim
// ---------------------------------------------------------------------------
__global__ void k_intra(const float* __restrict__ feat,
                        const int* __restrict__ inst,
                        const float* __restrict__ invnorm,
                        const int* __restrict__ icounts,
                        const int* __restrict__ offs,
                        const int* __restrict__ rank,
                        const int* __restrict__ list,
                        float* __restrict__ intra_sum) {
  __shared__ float red[256];
  int pix = blockIdx.x * blockDim.x + threadIdx.x;
  int b = pix >> 16;
  int n = pix & (HW_ - 1);
  int m = inst[pix];
  int cnt = icounts[b * K_ + m];
  float contrib = 0.f;
  if (m > 0 && cnt >= 2) {
    uint32_t h = (uint32_t)(b * K_ + m) * 0x9E3779B9u;
    h ^= h >> 16; h *= 0x85EBCA6Bu; h ^= h >> 13;
    int step = 1 + (int)(h % (uint32_t)(cnt - 1));
    int r = rank[pix];
    int p = list[(size_t)b * HW_ + offs[b * K_ + m] + ((r + step) % cnt)];
    const float* fn = feat + (size_t)b * C_ * HW_ + n;
    const float* fq = feat + (size_t)b * C_ * HW_ + p;
    float dot = 0.f;
#pragma unroll 4
    for (int c = 0; c < C_; ++c) {
      __builtin_prefetch(fn + (size_t)(c + 24) * HW_, 0, 1);
      __builtin_prefetch(fq + (size_t)(c + 24) * HW_, 0, 1);
      dot = fmaf(fn[(size_t)c * HW_], fq[(size_t)c * HW_], dot);
    }
    contrib = 1.0f - dot * invnorm[pix] * invnorm[b * HW_ + p];
  }
  red[threadIdx.x] = contrib;
  __syncthreads();
  for (int s2 = 128; s2 > 0; s2 >>= 1) {
    if (threadIdx.x < s2) red[threadIdx.x] += red[threadIdx.x + s2];
    __syncthreads();
  }
  if (threadIdx.x == 0) atomicAdd(&intra_sum[b], red[0]);
}

// ---------------------------------------------------------------------------
// K6: prototypes: mean, normalize, mask invalid, convert to f16
// ---------------------------------------------------------------------------
__global__ void k_proto(const float* __restrict__ segsum,
                        const int* __restrict__ icounts,
                        _Float16* __restrict__ proto) {
  __shared__ float red[64];
  int bs = blockIdx.x;                 // b*K_+s
  int s  = bs & (K_ - 1);
  int cnt = icounts[bs];
  float denom = fmaxf((float)cnt, 1.0f);
  const float* src = segsum + (size_t)bs * C_;
  float v[8], ss = 0.f;
#pragma unroll
  for (int i = 0; i < 8; ++i) {
    float x = src[threadIdx.x * 8 + i] / denom;
    v[i] = x;
    ss = fmaf(x, x, ss);
  }
  red[threadIdx.x] = ss;
  __syncthreads();
  for (int st = 32; st > 0; st >>= 1) {
    if (threadIdx.x < st) red[threadIdx.x] += red[threadIdx.x + st];
    __syncthreads();
  }
  float norm  = fmaxf(sqrtf(red[0]), EPS_);
  float scale = (cnt >= 2 && s > 0) ? (1.0f / norm) : 0.0f;   // zero invalid rows
  _Float16* dst = proto + (size_t)bs * C_;
#pragma unroll
  for (int i = 0; i < 8; ++i) dst[threadIdx.x * 8 + i] = (_Float16)(v[i] * scale);
}

// ---------------------------------------------------------------------------
// K7: Gram matrix P@P^T (64x64, K=512) via v_wmma_f32_16x16x32_f16 + hinge.
//     Staging: async global->LDS copy (ASYNCcnt path), then WMMA from LDS.
//     Invalid prototypes are zero rows -> hinge self-masks; skip diagonal.
// ---------------------------------------------------------------------------
__global__ void k_gram_inter(const _Float16* __restrict__ proto,
                             float* __restrict__ inter_sum) {
  extern __shared__ char smemraw[];
  _Float16* sp = (_Float16*)smemraw;                 // 64 x 512 halves = 64 KB
  __shared__ float red[128];
  int b = blockIdx.x;

  // ---- async global -> LDS staging: 65536 B, 128 threads * 16 B = 2 KB/pass
  {
    uint32_t lbase = (uint32_t)(uintptr_t)(void*)sp;
    const char* gbase = (const char*)(proto + (size_t)b * K_ * C_);
#pragma unroll 4
    for (int it = 0; it < 32; ++it) {
      uint32_t loff = lbase + (uint32_t)(it * 2048 + threadIdx.x * 16);
      const void* ga = gbase + it * 2048 + threadIdx.x * 16;
      asm volatile("global_load_async_to_lds_b128 %0, %1, off"
                   :: "v"(loff), "v"(ga)
                   : "memory");
    }
    asm volatile("s_wait_asynccnt 0x0" ::: "memory");
  }
  __syncthreads();

  int wave = threadIdx.x >> 5;          // M tile
  int lane = threadIdx.x & 31;
  int rowA  = (wave << 4) + (lane & 15);
  int kbase = (lane >> 4) << 3;         // lanes 16-31 hold K+8 halves
  float local = 0.f;

  for (int tn = 0; tn < 4; ++tn) {
    int rowB = (tn << 4) + (lane & 15); // B = P^T -> B column = P row
    v8f acc = {};
    for (int kk = 0; kk < C_; kk += 32) {
      v16h a, bm;
#pragma unroll
      for (int j = 0; j < 16; ++j) {
        int k = kk + ((j >> 3) << 4) + kbase + (j & 7);
        a[j]  = sp[rowA * C_ + k];
        bm[j] = sp[rowB * C_ + k];
      }
      acc = __builtin_amdgcn_wmma_f32_16x16x32_f16(
          /*neg_a=*/false, a, /*neg_b=*/false, bm,
          /*c_mod=*/(short)0, acc, /*reuse_a=*/false, /*reuse_b=*/false);
    }
    // D layout: VGPR r, lanes 0-15 -> M=r; lanes 16-31 -> M=r+8; N=lane%16
#pragma unroll
    for (int r = 0; r < 8; ++r) {
      int row = (wave << 4) + r + ((lane >> 4) << 3);
      int col = (tn << 4) + (lane & 15);
      if (row != col) local += fmaxf(acc[r] - MARG_, 0.0f);
    }
  }
  red[threadIdx.x] = local;
  __syncthreads();
  for (int st = 64; st > 0; st >>= 1) {
    if (threadIdx.x < st) red[threadIdx.x] += red[threadIdx.x + st];
    __syncthreads();
  }
  if (threadIdx.x == 0) inter_sum[b] = red[0];
}

// ---------------------------------------------------------------------------
// K8: final scalar
// ---------------------------------------------------------------------------
__global__ void k_final(const float* __restrict__ intra_sum,
                        const float* __restrict__ inter_sum,
                        const int* __restrict__ icounts,
                        float* __restrict__ out) {
  if (threadIdx.x == 0 && blockIdx.x == 0) {
    float ia = 0.f, ie = 0.f;
    for (int b = 0; b < B_; ++b) {
      float nvalid = (float)(HW_ - icounts[b * K_ + 0]);
      float intra = (nvalid >= 2.0f) ? intra_sum[b] / fmaxf(nvalid, 1.0f) : 0.f;
      int nproto = 0;
      for (int s = 1; s < K_; ++s)
        if (icounts[b * K_ + s] >= 2) nproto++;
      float npair = (float)(nproto * (nproto - 1));
      float inter = (nproto >= 2) ? inter_sum[b] / fmaxf(npair, 1.0f) : 0.f;
      ia += intra;
      ie += inter;
    }
    out[0] = ia / (float)B_ + ie / (float)B_;
  }
}

// ---------------------------------------------------------------------------
// Workspace layout (bytes):
//   0        invnorm  f32[B*HW]    2097152
//   2097152  rank     i32[B*HW]    2097152
//   4194304  list     i32[B*HW]    2097152
//   6291456  segsum   f32[B*K*C]   1048576
//   7340032  proto    f16[B*K*C]    524288
//   7864320  icounts  i32[B*K]        2048
//   7866368  offs     i32[B*K]        2048
//   7868416  cursor   i32[B*K]        2048
//   7870464  intra_sum f32[B] ; +64: inter_sum f32[B]
// ---------------------------------------------------------------------------
extern "C" void kernel_launch(void* const* d_in, const int* in_sizes, int n_in,
                              void* d_out, int out_size, void* d_ws, size_t ws_size,
                              hipStream_t stream) {
  (void)in_sizes; (void)n_in; (void)out_size; (void)ws_size;
  const float* feat = (const float*)d_in[0];
  const int*   inst = (const int*)d_in[1];
  char* ws = (char*)d_ws;

  float*    invnorm   = (float*)(ws + 0);
  int*      rank      = (int*)(ws + 2097152);
  int*      list      = (int*)(ws + 4194304);
  float*    segsum    = (float*)(ws + 6291456);
  _Float16* proto     = (_Float16*)(ws + 7340032);
  int*      icounts   = (int*)(ws + 7864320);
  int*      offs      = (int*)(ws + 7866368);
  int*      cursor    = (int*)(ws + 7868416);
  float*    intra_sum = (float*)(ws + 7870464);
  float*    inter_sum = (float*)(ws + 7870464 + 64);

  // zero accumulators (segsum; icounts/offs/cursor/sums)
  hipMemsetAsync(ws + 6291456, 0, 1048576, stream);
  hipMemsetAsync(ws + 7864320, 0, 3 * 2048 + 128, stream);

  k_norm_count <<<512, 256, 0, stream>>>(feat, inst, invnorm, icounts);
  k_offsets    <<<B_, 64, 0, stream>>>(icounts, offs);
  k_build_lists<<<2048, 256, 0, stream>>>(inst, offs, cursor, rank, list);
  k_segsum     <<<2048, 256, 0, stream>>>(feat, inst, invnorm, segsum);
  k_intra      <<<2048, 256, 0, stream>>>(feat, inst, invnorm, icounts, offs, rank, list, intra_sum);
  k_proto      <<<B_ * K_, 64, 0, stream>>>(segsum, icounts, proto);
  k_gram_inter <<<B_, 128, 65536, stream>>>(proto, inter_sum);
  k_final      <<<1, 64, 0, stream>>>(intra_sum, inter_sum, icounts, (float*)d_out);
}
